// GPRGNN_9414568312941
// MI455X (gfx1250) — matrix operand order repaired and checked by reading, not compiled, not run on hardware
//
#include <hip/hip_runtime.h>
#include <hip/hip_bf16.h>

// ---------- CDNA5 WMMA types ----------
typedef __attribute__((ext_vector_type(16))) __bf16 v16bf;
typedef __attribute__((ext_vector_type(8)))  float  v8f;

struct B32x2 { uint4 a, b; };   // 32 bytes == one v16bf fragment

static __device__ __forceinline__ v16bf load16bf(const void* p) {
  B32x2 t;
  t.a = *(const uint4*)p;
  t.b = *(const uint4*)((const char*)p + 16);
  return __builtin_bit_cast(v16bf, t);
}

// ---------- weight prep: fp32 row-major -> bf16 column-major ----------
// W1 [512,256] -> W1cm [256][512] ; W2 [256,16] -> W2cm [16][256]
__global__ void convert_weights(const float* __restrict__ W1,
                                const float* __restrict__ W2,
                                __bf16* __restrict__ W1cm,
                                __bf16* __restrict__ W2cm) {
  int t = blockIdx.x * blockDim.x + threadIdx.x;
  if (t < 512 * 256) {
    int k = t >> 8, c = t & 255;
    W1cm[c * 512 + k] = (__bf16)W1[t];
  }
  if (t < 256 * 16) {
    int k = t >> 4, c = t & 15;
    W2cm[c * 256 + k] = (__bf16)W2[t];
  }
}

// ---------- fused MLP: relu(x@W1+b1)@W2+b2 -> H0 ; out = pw[0]*H0 ----------
// One workgroup (256 thr = 8 waves) handles 16 rows x all 256 hidden cols.
// Phase 1: wave w owns hidden cols [32w,32w+32) as two 16x16 f32 accumulators.
//   K=512 consumed as 8 superchunks of 64 (4 WMMAs each), double-buffered in
//   LDS so staging of chunk c+1 overlaps WMMA on chunk c; 1 barrier per chunk.
// Phase 2: hidden tile (16x256 bf16 in LDS) -> each wave one K-chunk partial
//   of the single 16x16 output tile; reduced through LDS (no atomics).
__global__ __launch_bounds__(256)
void mlp_fused(const float* __restrict__ x, const __bf16* __restrict__ W1cm,
               const float* __restrict__ b1, const __bf16* __restrict__ W2cm,
               const float* __restrict__ b2, const float* __restrict__ pw,
               float* __restrict__ H0, float* __restrict__ out, int N) {
  __shared__ __align__(16) __bf16 sA[2][16 * 64];  // x staging, 2 x 2 KB
  __shared__ __align__(16) __bf16 sH[16 * 256];    // hidden tile, 8 KB
  __shared__ __align__(16) float  sP[8 * 256];     // gemm2 partials, 8 KB

  const int tid  = threadIdx.x;
  const int wave = tid >> 5;
  const int lane = tid & 31;
  const int lrow = lane & 15;   // row (A) / col (B,C) within tile
  const int lgrp = lane >> 4;   // lane group (K half)
  const int rowBase = blockIdx.x * 16;
  const int colBase = wave * 32;

  // staging coords: thread handles 4 consecutive K of one row (b128 load)
  const int sr  = (tid * 4) >> 6;    // row 0..15
  const int skk = (tid * 4) & 63;    // K offset within superchunk
  int srow = rowBase + sr; if (srow >= N) srow = N - 1;
  const float* xrow = x + (size_t)srow * 512 + skk;

  auto stage = [&](int buf, int k0) {
    float4 v = *(const float4*)(xrow + k0);
    __bf16* dst = &sA[buf][sr * 64 + skk];
    dst[0] = (__bf16)v.x; dst[1] = (__bf16)v.y;
    dst[2] = (__bf16)v.z; dst[3] = (__bf16)v.w;
  };

  v8f acc0 = {}; v8f acc1 = {};

  stage(0, 0);
  __syncthreads();

  for (int c = 0; c < 8; ++c) {
    const int cur = c & 1;
    if (c + 1 < 8) stage(cur ^ 1, (c + 1) * 64);   // overlap with WMMAs below

    const char* abase = (const char*)&sA[cur][0] + lrow * 128 + lgrp * 32;
    #pragma unroll
    for (int j = 0; j < 2; ++j) {
      // A fragment, contiguous K order: K in [16g,16g+16)
      B32x2 ta;
      ta.a = *(const uint4*)(abase + j * 64);
      ta.b = *(const uint4*)(abase + j * 64 + 16);
      v16bf afrag = __builtin_bit_cast(v16bf, ta);
      int k0 = c * 64 + j * 32;
      v16bf bf0 = load16bf(W1cm + (size_t)(colBase + lrow)      * 512 + k0 + lgrp * 16);
      v16bf bf1 = load16bf(W1cm + (size_t)(colBase + 16 + lrow) * 512 + k0 + lgrp * 16);
      acc0 = __builtin_amdgcn_wmma_f32_16x16x32_bf16(false, afrag, false, bf0,
                                                     (short)0, acc0, false, false);
      acc1 = __builtin_amdgcn_wmma_f32_16x16x32_bf16(false, afrag, false, bf1,
                                                     (short)0, acc1, false, false);
    }
    __syncthreads();
  }

  // bias + relu, write hidden tile to LDS as bf16
  {
    float bb0 = b1[colBase + lrow];
    float bb1 = b1[colBase + 16 + lrow];
    #pragma unroll
    for (int v = 0; v < 8; ++v) {
      int m = v + 8 * lgrp;                  // C layout: VGPR v -> row v or v+8
      float f0 = acc0[v] + bb0; f0 = f0 > 0.f ? f0 : 0.f;
      float f1 = acc1[v] + bb1; f1 = f1 > 0.f ? f1 : 0.f;
      sH[m * 256 + colBase + lrow]      = (__bf16)f0;
      sH[m * 256 + colBase + 16 + lrow] = (__bf16)f1;
    }
  }
  __syncthreads();

  // GEMM2: wave w computes partial over K in [32w, 32w+32)
  v8f acc2 = {};
  {
    int k0 = wave * 32;
    B32x2 ta;
    ta.a = *(const uint4*)((const char*)sH + lrow * 512 + (k0 + lgrp * 16) * 2);
    ta.b = *(const uint4*)((const char*)sH + lrow * 512 + (k0 + lgrp * 16 + 8) * 2);
    v16bf afrag = __builtin_bit_cast(v16bf, ta);
    v16bf bfrag = load16bf(W2cm + (size_t)lrow * 256 + k0 + lgrp * 16);
    acc2 = __builtin_amdgcn_wmma_f32_16x16x32_bf16(false, afrag, false, bfrag,
                                                   (short)0, acc2, false, false);
  }
  #pragma unroll
  for (int v = 0; v < 8; ++v) {
    int m = v + 8 * lgrp;
    sP[wave * 256 + m * 16 + lrow] = acc2[v];
  }
  __syncthreads();

  // reduce 8 wave-partials, add b2, emit H0 and out = pw[0]*H0
  {
    int m = tid >> 4, n = tid & 15;
    float s = 0.f;
    #pragma unroll
    for (int w = 0; w < 8; ++w) s += sP[w * 256 + m * 16 + n];
    s += b2[n];
    int grow = rowBase + m;
    if (grow < N) {
      H0[(size_t)grow * 16 + n]  = s;
      out[(size_t)grow * 16 + n] = pw[0] * s;
    }
  }
}

// ---------- sparse propagation ----------
__global__ void zero_f32(float* __restrict__ p, long long n) {
  long long i = (long long)blockIdx.x * blockDim.x + threadIdx.x;
  if (i < n) p[i] = 0.f;
}

__global__ void deg_count(const long long* __restrict__ row,
                          float* __restrict__ deg, long long E) {
  long long e = (long long)blockIdx.x * blockDim.x + threadIdx.x;
  if (e < E) unsafeAtomicAdd(&deg[row[e]], 1.0f);
}

__global__ void dinv_kernel(const float* __restrict__ deg,
                            float* __restrict__ dinv, int N) {
  int i = blockIdx.x * blockDim.x + threadIdx.x;
  if (i < N) { float d = deg[i]; dinv[i] = d > 0.f ? rsqrtf(d) : 0.f; }
}

__global__ void norm_kernel(const long long* __restrict__ row,
                            const long long* __restrict__ col,
                            const float* __restrict__ dinv,
                            float* __restrict__ nrm, long long E) {
  long long e = (long long)blockIdx.x * blockDim.x + threadIdx.x;
  if (e < E) nrm[e] = dinv[row[e]] * dinv[col[e]];
}

// one thread per (edge, channel): contiguous 64B gather/scatter per edge,
// L2-resident H; scatter via global_atomic_add_f32
__global__ __launch_bounds__(256)
void prop_scatter(const long long* __restrict__ row,
                  const long long* __restrict__ col,
                  const float* __restrict__ nrm,
                  const float* __restrict__ H, float* __restrict__ Hn,
                  long long E) {
  long long idx = (long long)blockIdx.x * blockDim.x + threadIdx.x;
  long long e = idx >> 4;
  int c = (int)(idx & 15);
  if (e < E) {
    float v = nrm[e] * H[row[e] * 16 + c];
    unsafeAtomicAdd(&Hn[col[e] * 16 + c], v);
  }
}

__global__ void axpy_kernel(const float* __restrict__ pw, int k,
                            const float* __restrict__ Hn,
                            float* __restrict__ out, long long n) {
  long long i = (long long)blockIdx.x * blockDim.x + threadIdx.x;
  if (i < n) out[i] += pw[k] * Hn[i];
}

// ---------- host launcher ----------
extern "C" void kernel_launch(void* const* d_in, const int* in_sizes, int n_in,
                              void* d_out, int out_size, void* d_ws, size_t ws_size,
                              hipStream_t stream) {
  (void)n_in; (void)out_size; (void)ws_size;
  const float*     x  = (const float*)d_in[0];
  const float*     W1 = (const float*)d_in[1];
  const float*     b1 = (const float*)d_in[2];
  const float*     W2 = (const float*)d_in[3];
  const float*     b2 = (const float*)d_in[4];
  const float*     pw = (const float*)d_in[5];
  const long long* ei = (const long long*)d_in[6];   // int64 [2,E]

  const int       N = in_sizes[0] / 512;
  const long long E = (long long)in_sizes[6] / 2;
  const long long* row = ei;
  const long long* col = ei + E;

  char* wsp = (char*)d_ws;
  auto alloc = [&](size_t bytes) {
    char* p = wsp;
    wsp += (bytes + 255) & ~(size_t)255;
    return p;
  };
  float*  Ha   = (float*)alloc((size_t)N * 16 * 4);
  float*  Hb   = (float*)alloc((size_t)N * 16 * 4);
  float*  deg  = (float*)alloc((size_t)N * 4);
  float*  dinv = (float*)alloc((size_t)N * 4);
  float*  nrm  = (float*)alloc((size_t)E * 4);
  __bf16* W1cm = (__bf16*)alloc((size_t)512 * 256 * 2);
  __bf16* W2cm = (__bf16*)alloc((size_t)256 * 16 * 2);
  float*  out  = (float*)d_out;

  convert_weights<<<(512 * 256 + 255) / 256, 256, 0, stream>>>(W1, W2, W1cm, W2cm);

  int nblk = (N + 15) / 16;
  mlp_fused<<<nblk, 256, 0, stream>>>(x, W1cm, b1, W2cm, b2, pw, Ha, out, N);

  zero_f32<<<(N + 255) / 256, 256, 0, stream>>>(deg, N);
  deg_count<<<(int)((E + 255) / 256), 256, 0, stream>>>(row, deg, E);
  dinv_kernel<<<(N + 255) / 256, 256, 0, stream>>>(deg, dinv, N);
  norm_kernel<<<(int)((E + 255) / 256), 256, 0, stream>>>(row, col, dinv, nrm, E);

  float* H  = Ha;
  float* Hn = Hb;
  const long long n16 = (long long)N * 16;
  for (int k = 1; k <= 10; ++k) {
    zero_f32<<<(int)((n16 + 255) / 256), 256, 0, stream>>>(Hn, n16);
    long long work = E * 16;
    prop_scatter<<<(int)((work + 255) / 256), 256, 0, stream>>>(row, col, nrm, H, Hn, E);
    axpy_kernel<<<(int)((n16 + 255) / 256), 256, 0, stream>>>(pw, k, Hn, out, n16);
    float* t = H; H = Hn; Hn = t;
  }
}
